// SimplifiedScaledDotProductAttention_35768487641139
// MI455X (gfx1250) — matrix operand order repaired and checked by reading, not compiled, not run on hardware
//
#include <hip/hip_runtime.h>

#define B_  2
#define S_  4096
#define D_  512
#define H_  8
#define DK_ 64

typedef __bf16 bf16_t;
typedef bf16_t v16bf __attribute__((ext_vector_type(16)));
typedef float  v8f   __attribute__((ext_vector_type(8)));

union FragA { v16bf v; unsigned short u[16]; };

__device__ __forceinline__ unsigned short f2bf(float f) {
  return __builtin_bit_cast(unsigned short, (__bf16)f);
}
__device__ __forceinline__ unsigned int pack2bf(float x, float y) {
  return (unsigned int)f2bf(x) | ((unsigned int)f2bf(y) << 16);
}

__device__ __forceinline__ v8f wmma_bf16(v16bf a, v16bf b, v8f c) {
  return __builtin_amdgcn_wmma_f32_16x16x32_bf16(false, a, false, b,
                                                 (short)0, c, false, false);
}

// ---------------------------------------------------------------------------
// Kernel 1: flash attention. One (b,h) + 64 query rows per block (4 waves),
// 16 query rows per wave, 64 keys per iteration. K/V staged in LDS as bf16.
// 1/sqrt(dk) folded into Q. Row sums computed with a ones-matrix WMMA.
// 18 WMMAs per iteration between barriers.
// ---------------------------------------------------------------------------
__global__ __launch_bounds__(128)
void attn_kernel(const float* __restrict__ Q, const float* __restrict__ K,
                 const float* __restrict__ V, unsigned short* __restrict__ O) {
  __shared__ unsigned short sK[64 * 64];    // 8 KB
  __shared__ unsigned short sV[64 * 64];    // 8 KB
  __shared__ float sP[4 * 16 * 64];         // 16 KB; per-wave P scratch (reused for Q staging)

  const int tid  = threadIdx.x;
  const int wave = tid >> 5;
  const int lane = tid & 31;
  const int half = lane >> 4;
  const int l16  = lane & 15;

  const int qblk  = blockIdx.x & 63;        // S/64 q-tiles per (b,h)
  const int bh    = blockIdx.x >> 6;
  const int b     = bh >> 3;                // H = 8
  const int h     = bh & 7;
  const int qbase = qblk * 64;

  // ---- stage Q tile (64 x 64) * (1/8) as bf16 into sP region ----
  unsigned short* sQ = (unsigned short*)sP;
#pragma unroll
  for (int j = 0; j < 8; ++j) {
    int idx = tid + 128 * j;                // 0..1023 float4 chunks
    int row = idx >> 4;
    int c4  = idx & 15;
    float4 f = *(const float4*)(Q + (size_t)(b * S_ + qbase + row) * D_ + h * DK_ + c4 * 4);
    uint2 p;
    p.x = pack2bf(f.x * 0.125f, f.y * 0.125f);
    p.y = pack2bf(f.z * 0.125f, f.w * 0.125f);
    *(uint2*)&sQ[row * 64 + c4 * 4] = p;
  }
  __syncthreads();

  // per-wave Q A-fragments (rows wave*16 .. +15), K-halves 0..31 / 32..63
  FragA qa_lo, qa_hi;
#pragma unroll
  for (int i = 0; i < 16; ++i) {
    int kk = (half << 4) + i;
    qa_lo.u[i] = sQ[(wave * 16 + l16) * 64 + kk];
    qa_hi.u[i] = sQ[(wave * 16 + l16) * 64 + 32 + kk];
  }

  // constant all-ones B fragment (32x16 of bf16 1.0) for row-sum WMMA
  FragA ones;
#pragma unroll
  for (int i = 0; i < 16; ++i) ones.u[i] = 0x3f80;

  v8f acc0 = {}, acc1 = {}, acc2 = {}, acc3 = {};
  v8f accL = {};                            // running softmax denominators
  float mrow[8];
#pragma unroll
  for (int r = 0; r < 8; ++r) mrow[r] = -1e30f;

  float* sPw = sP + wave * (16 * 64);

  for (int kt = 0; kt < S_ / 64; ++kt) {
    const int kbase = kt * 64;
    __syncthreads();                        // prev iteration done with sK/sV
    // ---- stage K,V tiles (64 keys x 64 dims) as bf16 ----
#pragma unroll
    for (int j = 0; j < 8; ++j) {
      int idx = tid + 128 * j;              // 0..1023 float4 chunks
      int row = idx >> 4;
      int c4  = idx & 15;
      size_t g = (size_t)(b * S_ + kbase + row) * D_ + h * DK_ + c4 * 4;
      float4 fk = *(const float4*)(K + g);
      float4 fv = *(const float4*)(V + g);
      uint2 pk, pv;
      pk.x = pack2bf(fk.x, fk.y);  pk.y = pack2bf(fk.z, fk.w);
      pv.x = pack2bf(fv.x, fv.y);  pv.y = pack2bf(fv.z, fv.w);
      *(uint2*)&sK[row * 64 + c4 * 4] = pk;
      *(uint2*)&sV[row * 64 + c4 * 4] = pv;
    }
    __syncthreads();

    // ---- S = (Q/8) * K^T : four 16-key column fragments ----
    v8f cf[4];
#pragma unroll
    for (int nf = 0; nf < 4; ++nf) {
      v8f c = {};
      FragA bk;
#pragma unroll
      for (int i = 0; i < 16; ++i) bk.u[i] = sK[(nf * 16 + l16) * 64 + half * 16 + i];
      c = wmma_bf16(qa_lo.v, bk.v, c);
#pragma unroll
      for (int i = 0; i < 16; ++i) bk.u[i] = sK[(nf * 16 + l16) * 64 + 32 + half * 16 + i];
      c = wmma_bf16(qa_hi.v, bk.v, c);
      cf[nf] = c;
    }

    // ---- online softmax: one max butterfly per row covers all 64 keys ----
#pragma unroll
    for (int r = 0; r < 8; ++r) {
      float mx = fmaxf(fmaxf(cf[0][r], cf[1][r]), fmaxf(cf[2][r], cf[3][r]));
#pragma unroll
      for (int m = 1; m <= 8; m <<= 1) mx = fmaxf(mx, __shfl_xor(mx, m, 32));
      float newm = fmaxf(mrow[r], mx);
      float corr = __expf(mrow[r] - newm);
      mrow[r] = newm;
      acc0[r] *= corr; acc1[r] *= corr; acc2[r] *= corr; acc3[r] *= corr;
      accL[r] *= corr;
      int mi = r + half * 8;
      sPw[mi * 64 +      l16] = __expf(cf[0][r] - newm);
      sPw[mi * 64 + 16 + l16] = __expf(cf[1][r] - newm);
      sPw[mi * 64 + 32 + l16] = __expf(cf[2][r] - newm);
      sPw[mi * 64 + 48 + l16] = __expf(cf[3][r] - newm);
    }
    asm volatile("s_wait_dscnt 0" ::: "memory");   // P stores visible within wave

    // ---- P A-fragments: keys 0..31 and 32..63 ----
    FragA pa0, pa1;
#pragma unroll
    for (int i = 0; i < 16; ++i) {
      int kk = (half << 4) + i;
      pa0.u[i] = f2bf(sPw[l16 * 64 + kk]);
      pa1.u[i] = f2bf(sPw[l16 * 64 + 32 + kk]);
    }

    // ---- O += P * V  (and row sums via ones-matrix WMMA) ----
    FragA vb;
#pragma unroll
    for (int nf = 0; nf < 4; ++nf) {
      v8f a = (nf == 0) ? acc0 : (nf == 1) ? acc1 : (nf == 2) ? acc2 : acc3;
#pragma unroll
      for (int i = 0; i < 16; ++i) vb.u[i] = sV[(half * 16 + i) * 64 + nf * 16 + l16];
      a = wmma_bf16(pa0.v, vb.v, a);
#pragma unroll
      for (int i = 0; i < 16; ++i) vb.u[i] = sV[(32 + half * 16 + i) * 64 + nf * 16 + l16];
      a = wmma_bf16(pa1.v, vb.v, a);
      if (nf == 0) acc0 = a; else if (nf == 1) acc1 = a; else if (nf == 2) acc2 = a; else acc3 = a;
    }
    accL = wmma_bf16(pa0.v, ones.v, accL);
    accL = wmma_bf16(pa1.v, ones.v, accL);
  }

  // ---- normalize, emit bf16 attention output to workspace ----
#pragma unroll
  for (int r = 0; r < 8; ++r) {
    int row = qbase + wave * 16 + r + half * 8;
    float inv = 1.0f / accL[r];
    size_t base = (size_t)(b * S_ + row) * D_ + h * DK_;
    O[base +      l16] = f2bf(acc0[r] * inv);
    O[base + 16 + l16] = f2bf(acc1[r] * inv);
    O[base + 32 + l16] = f2bf(acc2[r] * inv);
    O[base + 48 + l16] = f2bf(acc3[r] * inv);
  }
}

// ---------------------------------------------------------------------------
// Kernel 2: Out[8192x512] = A_bf16[8192x512] * W[512x512] + bias.
// 64x64 block tile, 4 waves, each wave 16x64, K staged 64-wide as bf16.
// 8 WMMAs per K-step between barriers.
// ---------------------------------------------------------------------------
__global__ __launch_bounds__(128)
void proj_kernel(const unsigned short* __restrict__ A, const float* __restrict__ W,
                 const float* __restrict__ bias, float* __restrict__ Out) {
  __shared__ unsigned short sA[64 * 64];    // 8 KB
  __shared__ unsigned short sB[64 * 64];    // 8 KB

  const int tid  = threadIdx.x;
  const int wave = tid >> 5;
  const int lane = tid & 31;
  const int half = lane >> 4;
  const int l16  = lane & 15;

  const int ntile = blockIdx.x & 7;         // D/64 = 8
  const int mtile = blockIdx.x >> 3;        // 8192/64 = 128

  v8f acc0 = {}, acc1 = {}, acc2 = {}, acc3 = {};

  for (int kk = 0; kk < D_ / 64; ++kk) {
    const int k0 = kk * 64;
    __syncthreads();
    // stage A tile (64x64 bf16) straight from workspace
#pragma unroll
    for (int j = 0; j < 4; ++j) {
      int c    = tid + 128 * j;             // 0..511 chunks of 8 bf16
      int row  = c >> 3;
      int colc = (c & 7) * 8;
      *(uint4*)&sA[row * 64 + colc] =
          *(const uint4*)(A + (size_t)(mtile * 64 + row) * D_ + k0 + colc);
    }
    // stage W tile (64x64), fp32 -> bf16 via native convert
#pragma unroll
    for (int j = 0; j < 8; ++j) {
      int idx = tid + 128 * j;              // 0..1023 float4 chunks
      int row = idx >> 4;
      int c4  = idx & 15;
      float4 f = *(const float4*)(W + (size_t)(k0 + row) * D_ + ntile * 64 + c4 * 4);
      uint2 p;
      p.x = pack2bf(f.x, f.y);
      p.y = pack2bf(f.z, f.w);
      *(uint2*)&sB[row * 64 + c4 * 4] = p;
    }
    __syncthreads();

    FragA a_lo, a_hi;
#pragma unroll
    for (int i = 0; i < 16; ++i) {
      int kidx = (half << 4) + i;
      a_lo.u[i] = sA[(wave * 16 + l16) * 64 + kidx];
      a_hi.u[i] = sA[(wave * 16 + l16) * 64 + 32 + kidx];
    }
    FragA bb;
#pragma unroll
    for (int nf = 0; nf < 4; ++nf) {
      v8f a = (nf == 0) ? acc0 : (nf == 1) ? acc1 : (nf == 2) ? acc2 : acc3;
#pragma unroll
      for (int i = 0; i < 16; ++i) bb.u[i] = sB[(half * 16 + i) * 64 + nf * 16 + l16];
      a = wmma_bf16(a_lo.v, bb.v, a);
#pragma unroll
      for (int i = 0; i < 16; ++i) bb.u[i] = sB[(32 + half * 16 + i) * 64 + nf * 16 + l16];
      a = wmma_bf16(a_hi.v, bb.v, a);
      if (nf == 0) acc0 = a; else if (nf == 1) acc1 = a; else if (nf == 2) acc2 = a; else acc3 = a;
    }
  }

#pragma unroll
  for (int r = 0; r < 8; ++r) {
    int row = mtile * 64 + wave * 16 + r + half * 8;
    size_t base = (size_t)row * D_ + ntile * 64;
    Out[base +      l16] = acc0[r] + bias[ntile * 64 +      l16];
    Out[base + 16 + l16] = acc1[r] + bias[ntile * 64 + 16 + l16];
    Out[base + 32 + l16] = acc2[r] + bias[ntile * 64 + 32 + l16];
    Out[base + 48 + l16] = acc3[r] + bias[ntile * 64 + 48 + l16];
  }
}

extern "C" void kernel_launch(void* const* d_in, const int* in_sizes, int n_in,
                              void* d_out, int out_size, void* d_ws, size_t ws_size,
                              hipStream_t stream) {
  const float* Q  = (const float*)d_in[0];
  const float* K  = (const float*)d_in[1];
  const float* V  = (const float*)d_in[2];
  const float* W  = (const float*)d_in[3];
  const float* bo = (const float*)d_in[4];
  unsigned short* attn = (unsigned short*)d_ws;   // B*S*D bf16 = 8 MB

  attn_kernel<<<dim3(B_ * H_ * (S_ / 64)), dim3(128), 0, stream>>>(Q, K, V, attn);
  proj_kernel<<<dim3((B_ * S_ / 64) * (D_ / 64)), dim3(128), 0, stream>>>(attn, W, bo, (float*)d_out);
}
